// MultiHeadAttention_82695300317274
// MI455X (gfx1250) — compile-verified, matrix operand order checked by reading, not run on hardware
//
#include <hip/hip_runtime.h>
#include <hip/hip_bf16.h>

typedef __attribute__((ext_vector_type(16))) _Float16 v16h;
typedef __attribute__((ext_vector_type(8)))  _Float16 h8;
typedef __attribute__((ext_vector_type(4)))  _Float16 h4;
typedef __attribute__((ext_vector_type(8)))  float    v8f;

#define WMMA_F16(a, b, c) \
  __builtin_amdgcn_wmma_f32_16x16x32_f16(false, (a), false, (b), (short)0, (c), false, false)

#define B_ 2
#define L_ 2048
#define D_ 768
#define H_ 12
#define DK_ 64
#define NEGBIG (-3.0e38f)

#if defined(__has_builtin)
#if __has_builtin(__builtin_amdgcn_global_load_async_to_lds_b128)
#define HAVE_ASYNC_LDS 1
#endif
#endif

#ifdef HAVE_ASYNC_LDS
// Builtin prototype (from hipcc diagnostic): param0 = int4-vector ptr in AS1
// (__device__), param1 = int4-vector ptr in AS3 (__shared__), then imm offset, imm cpol.
typedef int i4_ __attribute__((vector_size(16)));
typedef __attribute__((address_space(1))) i4_* gp_v4i;
typedef __attribute__((address_space(3))) i4_* lp_v4i;
__device__ __forceinline__ void async_cp16(const void* g, void* l) {
  __builtin_amdgcn_global_load_async_to_lds_b128((gp_v4i)(void*)g, (lp_v4i)l, 0, 0);
}
#endif

__device__ __forceinline__ v16h mk16(h8 lo, h8 hi) {
  v16h r;
#pragma unroll
  for (int e = 0; e < 8; ++e) { r[e] = lo[e]; r[e + 8] = hi[e]; }
  return r;
}

// Load one 16-half A/B fragment (wave32 16x16x32 f16 layout): two contiguous
// 16-byte chunks.
__device__ __forceinline__ v16h ldfrag(const _Float16* row, int off) {
  return mk16(*(const h8*)(row + off), *(const h8*)(row + off + 16));
}

__device__ __forceinline__ float redmax16(float v) {
  v = fmaxf(v, __shfl_xor(v, 1, 32));
  v = fmaxf(v, __shfl_xor(v, 2, 32));
  v = fmaxf(v, __shfl_xor(v, 4, 32));
  v = fmaxf(v, __shfl_xor(v, 8, 32));
  return v;   // masks < 16 => stays within each 16-lane half
}
__device__ __forceinline__ float redsum16(float v) {
  v += __shfl_xor(v, 1, 32);
  v += __shfl_xor(v, 2, 32);
  v += __shfl_xor(v, 4, 32);
  v += __shfl_xor(v, 8, 32);
  return v;
}

// ---------------- Kernel 0a: x (f32) -> xh (f16), elementwise -----------------------
__global__ __launch_bounds__(256) void cvt_x_f16(const float* __restrict__ src,
                                                 _Float16* __restrict__ dst) {
  const size_t i = ((size_t)blockIdx.x * 256 + threadIdx.x) * 4;
  const float4 v = *(const float4*)(src + i);
  h4 o;
  o[0] = (_Float16)v.x; o[1] = (_Float16)v.y;
  o[2] = (_Float16)v.z; o[3] = (_Float16)v.w;
  *(h4*)(dst + i) = o;
}

// ---------------- Kernel 0b: W (rows x cols f32) -> Wt (cols x rows f16) ------------
__global__ __launch_bounds__(256) void cvt_transpose(const float* __restrict__ src,
                                                     _Float16* __restrict__ dst,
                                                     int rows, int cols) {
  __shared__ _Float16 tile[64][65];            // +1 pad: no bank conflicts
  const int c0 = blockIdx.x * 64, r0 = blockIdx.y * 64;
  const int tc = threadIdx.x & 63;
  const int t4 = threadIdx.x >> 6;             // 0..3
#pragma unroll
  for (int rr = 0; rr < 64; rr += 4)
    tile[rr + t4][tc] = (_Float16)src[(size_t)(r0 + rr + t4) * cols + c0 + tc];
  __syncthreads();
#pragma unroll
  for (int rr = 0; rr < 64; rr += 4)
    dst[(size_t)(c0 + rr + t4) * rows + r0 + tc] = tile[tc][rr + t4];
}

// ---------------- Kernel 1: QKV = xh @ Wqkv + bqkv -----------------------------------
// A = xh (4096x768 f16), B = Wt (2304x768 f16, N-major). One wave -> 16x64 strip.
// Explicit ping-pong double buffer (no register rotation moves).
__global__ __launch_bounds__(256) void qkv_gemm(const _Float16* __restrict__ xh,
                                                const _Float16* __restrict__ Wt,
                                                const float* __restrict__ bqkv,
                                                _Float16* __restrict__ qws,
                                                _Float16* __restrict__ kws,
                                                _Float16* __restrict__ vtws) {
  const int lane = threadIdx.x & 31;
  const int wave = threadIdx.x >> 5;
  const int NS   = (3 * D_) / 64;              // 36 strips
  const int task = blockIdx.x * 8 + wave;      // 1152*8 = 9216 tasks
  const int mt = task / NS, ns = task % NS;
  const int m0 = mt * 16, n0 = ns * 64;
  const int hi8 = (lane >> 4) << 3;
  const int l15 = lane & 15;

  v8f acc[4];
#pragma unroll
  for (int t = 0; t < 4; ++t)
#pragma unroll
    for (int j = 0; j < 8; ++j) acc[t][j] = 0.0f;

  const _Float16* xrow = xh + (size_t)(m0 + l15) * D_;
  const _Float16* wr[4];
  wr[0] = Wt + (size_t)(n0 + l15) * D_;
  wr[1] = wr[0] + (size_t)16 * D_;
  wr[2] = wr[0] + (size_t)32 * D_;
  wr[3] = wr[0] + (size_t)48 * D_;

  v16h aA = ldfrag(xrow, hi8);
  v16h bA[4];
#pragma unroll
  for (int t = 0; t < 4; ++t) bA[t] = ldfrag(wr[t], hi8);
  v16h aB, bB[4];

  for (int k0 = 0; k0 < D_ - 64; k0 += 64) {   // 11 iterations (24 k-steps total)
    const int k1 = k0 + 32 + hi8, k2 = k0 + 64 + hi8;
    aB = ldfrag(xrow, k1);
#pragma unroll
    for (int t = 0; t < 4; ++t) bB[t] = ldfrag(wr[t], k1);
#pragma unroll
    for (int t = 0; t < 4; ++t) acc[t] = WMMA_F16(aA, bA[t], acc[t]);
    aA = ldfrag(xrow, k2);
#pragma unroll
    for (int t = 0; t < 4; ++t) bA[t] = ldfrag(wr[t], k2);
#pragma unroll
    for (int t = 0; t < 4; ++t) acc[t] = WMMA_F16(aB, bB[t], acc[t]);
  }
  { // tail: buffers A hold k = D_-64; load final k = D_-32 into B
    const int k1 = D_ - 32 + hi8;
    aB = ldfrag(xrow, k1);
#pragma unroll
    for (int t = 0; t < 4; ++t) bB[t] = ldfrag(wr[t], k1);
#pragma unroll
    for (int t = 0; t < 4; ++t) acc[t] = WMMA_F16(aA, bA[t], acc[t]);
#pragma unroll
    for (int t = 0; t < 4; ++t) acc[t] = WMMA_F16(aB, bB[t], acc[t]);
  }

  const int part = n0 / D_;                    // 0=Q, 1=K, 2=V (strip is 64-aligned)
  const int h    = (n0 % D_) >> 6;
#pragma unroll
  for (int t = 0; t < 4; ++t) {
    const int dk = t * 16 + l15;
    const float bias = bqkv[n0 + t * 16 + l15];
#pragma unroll
    for (int j = 0; j < 8; ++j) {
      const int m    = m0 + j + hi8;
      const int bidx = m >> 11;
      const int lrow = m & (L_ - 1);
      const _Float16 val = (_Float16)(acc[t][j] + bias);
      const size_t bh = (size_t)(bidx * H_ + h);
      if (part == 0)       qws[(bh * L_ + lrow) * DK_ + dk] = val;
      else if (part == 1)  kws[(bh * L_ + lrow) * DK_ + dk] = val;
      else                 vtws[(bh * DK_ + dk) * L_ + lrow] = val;  // transposed
    }
  }
}

// ---------------- Kernel 2: flash attention, 64 query rows per block -----------------
__global__ __launch_bounds__(128) void attn(const _Float16* __restrict__ qws,
                                            const _Float16* __restrict__ kws,
                                            const _Float16* __restrict__ vtws,
                                            _Float16* __restrict__ yws) {
  __shared__ _Float16 lds_k[32 * DK_];         // [key][dk]
  __shared__ _Float16 lds_vt[DK_ * 32];        // [dk][key]
  __shared__ _Float16 lds_p[4][16 * 32];

  const int tid  = threadIdx.x;
  const int lane = tid & 31;
  const int wave = tid >> 5;
  const int nqb  = L_ / 64;
  const int bh   = blockIdx.x / nqb;
  const int qblk = blockIdx.x % nqb;
  const int b = bh / H_, h = bh % H_;
  const size_t head_base = (size_t)bh * L_ * DK_;   // same stride for k and vt
  const int q0  = qblk * 64 + wave * 16;
  const int hi8 = (lane >> 4) << 3;
  const int l15 = lane & 15;

  // Q fragments (16x64 tile in 2 A-fragments), b128 loads
  const _Float16* qp = qws + head_base + (size_t)(q0 + l15) * DK_;
  const v16h qa0 = ldfrag(qp, hi8);
  const v16h qa1 = ldfrag(qp, 32 + hi8);

  v8f O[4];
#pragma unroll
  for (int t = 0; t < 4; ++t)
#pragma unroll
    for (int j = 0; j < 8; ++j) O[t][j] = 0.0f;
  float m_i[8], l_i[8];
#pragma unroll
  for (int j = 0; j < 8; ++j) { m_i[j] = NEGBIG; l_i[j] = 0.0f; }

  const int nkb = qblk * 2 + 2;                // causal key-block bound
  for (int kb = 0; kb < nkb; ++kb) {
    // ---- stage K (32x64) and V^T (64x32) tiles into LDS ----
    const _Float16* ksrc  = kws  + head_base + (size_t)kb * 32 * DK_; // contiguous 4KB
    const _Float16* vtsrc = vtws + head_base + (size_t)kb * 32;       // rows stride L_
    const int vrow = tid >> 2, vq = (tid & 3) * 8;                    // 64 rows x 4 chunks
#ifdef HAVE_ASYNC_LDS
    async_cp16(ksrc + tid * 8,        &lds_k[tid * 8]);
    async_cp16(ksrc + 1024 + tid * 8, &lds_k[1024 + tid * 8]);
    async_cp16(vtsrc + (size_t)vrow * L_ + vq,        &lds_vt[vrow * 32 + vq]);
    async_cp16(vtsrc + (size_t)(vrow + 32) * L_ + vq, &lds_vt[(vrow + 32) * 32 + vq]);
    asm volatile("s_wait_asynccnt 0" ::: "memory");
#else
    *(h8*)&lds_k[tid * 8]        = *(const h8*)(ksrc + tid * 8);
    *(h8*)&lds_k[1024 + tid * 8] = *(const h8*)(ksrc + 1024 + tid * 8);
    *(h8*)&lds_vt[vrow * 32 + vq]        = *(const h8*)(vtsrc + (size_t)vrow * L_ + vq);
    *(h8*)&lds_vt[(vrow + 32) * 32 + vq] = *(const h8*)(vtsrc + (size_t)(vrow + 32) * L_ + vq);
#endif
    __syncthreads();

    if (kb + 1 < nkb) {
      __builtin_prefetch(ksrc + 32 * DK_ + lane * 4, 0, 1);
      __builtin_prefetch(vtsrc + 32 + lane, 0, 1);
    }

    const int kbg = kb * 32;
    if (kbg <= q0 + 15) {                      // wave-uniform; EXEC stays all-1s
      // ---- S = Q * K^T : two 16x16 tiles ----
      v8f S[2];
#pragma unroll
      for (int t = 0; t < 2; ++t) {
        const _Float16* kr = &lds_k[(t * 16 + l15) * DK_];
        const v16h kf0 = ldfrag(kr, hi8);
        const v16h kf1 = ldfrag(kr, 32 + hi8);
        v8f z;
#pragma unroll
        for (int j = 0; j < 8; ++j) z[j] = 0.0f;
        z = WMMA_F16(qa0, kf0, z);
        z = WMMA_F16(qa1, kf1, z);
        S[t] = z;
      }
      // ---- scale + analytic causal mask ----
#pragma unroll
      for (int t = 0; t < 2; ++t) {
        const int col = kbg + t * 16 + l15;
#pragma unroll
        for (int j = 0; j < 8; ++j) {
          const int row = q0 + j + hi8;
          const float s = S[t][j] * 0.125f;    // 1/sqrt(64)
          S[t][j] = (col <= row) ? s : NEGBIG;
        }
      }
      // ---- online softmax ----
      float alpha[8];
#pragma unroll
      for (int j = 0; j < 8; ++j) {
        float mn = redmax16(fmaxf(S[0][j], S[1][j]));
        mn = fmaxf(mn, m_i[j]);
        alpha[j] = __expf(m_i[j] - mn);
        m_i[j] = mn;
        S[0][j] = __expf(S[0][j] - mn);
        S[1][j] = __expf(S[1][j] - mn);
        l_i[j] = l_i[j] * alpha[j] + redsum16(S[0][j] + S[1][j]);
      }
#pragma unroll
      for (int t = 0; t < 4; ++t)
#pragma unroll
        for (int j = 0; j < 8; ++j) O[t][j] *= alpha[j];

      // ---- P: C-layout -> per-wave LDS -> A-fragment layout ----
#pragma unroll
      for (int t = 0; t < 2; ++t)
#pragma unroll
        for (int j = 0; j < 8; ++j)
          lds_p[wave][(j + hi8) * 32 + t * 16 + l15] = (_Float16)S[t][j];
      asm volatile("s_wait_dscnt 0" ::: "memory");
      const v16h pa = ldfrag(&lds_p[wave][l15 * 32], hi8);

      // ---- O(16x64) += P(16x32) * V(32x64), V read from V^T staging ----
#pragma unroll
      for (int t = 0; t < 4; ++t) {
        const v16h vb = ldfrag(&lds_vt[(t * 16 + l15) * 32], hi8);
        O[t] = WMMA_F16(pa, vb, O[t]);
      }
    }
    __syncthreads();
  }

  // epilogue: normalize and write y[b, l, h*64+dk] (f16)
#pragma unroll
  for (int j = 0; j < 8; ++j) {
    const float inv = 1.0f / l_i[j];
    const int row = q0 + j + hi8;
    _Float16* yr = yws + ((size_t)b * L_ + row) * D_ + h * DK_;
#pragma unroll
    for (int t = 0; t < 4; ++t)
      yr[t * 16 + l15] = (_Float16)(O[t][j] * inv);
  }
}

// ---------------- Kernel 3: out = y @ Wo + bo (fp32 out), ping-pong -----------------
__global__ __launch_bounds__(256) void out_gemm(const _Float16* __restrict__ y,
                                                const _Float16* __restrict__ Wot,
                                                const float* __restrict__ bo,
                                                float* __restrict__ out) {
  const int lane = threadIdx.x & 31;
  const int wave = threadIdx.x >> 5;
  const int NS   = D_ / 64;                    // 12 strips
  const int task = blockIdx.x * 8 + wave;      // 384*8 = 3072 tasks
  const int mt = task / NS, ns = task % NS;
  const int m0 = mt * 16, n0 = ns * 64;
  const int hi8 = (lane >> 4) << 3;
  const int l15 = lane & 15;

  v8f acc[4];
#pragma unroll
  for (int t = 0; t < 4; ++t)
#pragma unroll
    for (int j = 0; j < 8; ++j) acc[t][j] = 0.0f;

  const _Float16* yrow = y + (size_t)(m0 + l15) * D_;
  const _Float16* wr[4];
  wr[0] = Wot + (size_t)(n0 + l15) * D_;
  wr[1] = wr[0] + (size_t)16 * D_;
  wr[2] = wr[0] + (size_t)32 * D_;
  wr[3] = wr[0] + (size_t)48 * D_;

  v16h aA = ldfrag(yrow, hi8);
  v16h bA[4];
#pragma unroll
  for (int t = 0; t < 4; ++t) bA[t] = ldfrag(wr[t], hi8);
  v16h aB, bB[4];

  for (int k0 = 0; k0 < D_ - 64; k0 += 64) {
    const int k1 = k0 + 32 + hi8, k2 = k0 + 64 + hi8;
    aB = ldfrag(yrow, k1);
#pragma unroll
    for (int t = 0; t < 4; ++t) bB[t] = ldfrag(wr[t], k1);
#pragma unroll
    for (int t = 0; t < 4; ++t) acc[t] = WMMA_F16(aA, bA[t], acc[t]);
    aA = ldfrag(yrow, k2);
#pragma unroll
    for (int t = 0; t < 4; ++t) bA[t] = ldfrag(wr[t], k2);
#pragma unroll
    for (int t = 0; t < 4; ++t) acc[t] = WMMA_F16(aB, bB[t], acc[t]);
  }
  {
    const int k1 = D_ - 32 + hi8;
    aB = ldfrag(yrow, k1);
#pragma unroll
    for (int t = 0; t < 4; ++t) bB[t] = ldfrag(wr[t], k1);
#pragma unroll
    for (int t = 0; t < 4; ++t) acc[t] = WMMA_F16(aA, bA[t], acc[t]);
#pragma unroll
    for (int t = 0; t < 4; ++t) acc[t] = WMMA_F16(aB, bB[t], acc[t]);
  }

#pragma unroll
  for (int t = 0; t < 4; ++t) {
    const int n = n0 + t * 16 + l15;
    const float bias = bo[n];
#pragma unroll
    for (int j = 0; j < 8; ++j)
      out[(size_t)(m0 + j + hi8) * D_ + n] = acc[t][j] + bias;
  }
}

extern "C" void kernel_launch(void* const* d_in, const int* in_sizes, int n_in,
                              void* d_out, int out_size, void* d_ws, size_t ws_size,
                              hipStream_t stream) {
  const float* x    = (const float*)d_in[0];
  // d_in[1] = additive causal mask: computed analytically, never read.
  const float* Wqkv = (const float*)d_in[2];
  const float* bqkv = (const float*)d_in[3];
  const float* Wo   = (const float*)d_in[4];
  const float* bo   = (const float*)d_in[5];
  float* out = (float*)d_out;

  const size_t HEAD = (size_t)B_ * H_ * L_ * DK_;        // 3,145,728
  _Float16* qws  = (_Float16*)d_ws;
  _Float16* kws  = qws + HEAD;
  _Float16* vtws = kws + HEAD;                           // [b,h,dk,l]
  _Float16* yws  = vtws + HEAD;                          // 4096x768
  _Float16* xh   = yws + (size_t)B_ * L_ * D_;           // 4096x768
  _Float16* wtq  = xh + (size_t)B_ * L_ * D_;            // 2304x768
  _Float16* wto  = wtq + (size_t)(3 * D_) * D_;          // 768x768
  // total ws: ~36.2 MB of f16

  // one-time precision/layout conversion (compute-bound problem; this is ~1us of HBM)
  cvt_x_f16<<<(B_ * L_ * D_) / 1024, 256, 0, stream>>>(x, xh);
  cvt_transpose<<<dim3((3 * D_) / 64, D_ / 64), 256, 0, stream>>>(Wqkv, wtq, D_, 3 * D_);
  cvt_transpose<<<dim3(D_ / 64, D_ / 64), 256, 0, stream>>>(Wo, wto, D_, D_);

  qkv_gemm<<<1152, 256, 0, stream>>>(xh, wtq, bqkv, qws, kws, vtws);
  attn<<<B_ * H_ * (L_ / 64), 128, 0, stream>>>(qws, kws, vtws, yws);
  out_gemm<<<384, 256, 0, stream>>>(yws, wto, bo, out);
}
